// PNLTransitionPrior_43465069035862
// MI455X (gfx1250) — compile-verified, matrix operand order checked by reading, not run on hardware
//
#include <hip/hip_runtime.h>
#include <hip/hip_bf16.h>

typedef __attribute__((ext_vector_type(16))) _Float16 v16h;
typedef __attribute__((ext_vector_type(8)))  _Float16 v8h;
typedef __attribute__((ext_vector_type(4)))  _Float16 v4h;
typedef __attribute__((ext_vector_type(8)))  float    v8f;

#define TPB   256
#define WAVES 8
#define DD    64
#define HH    128
#define TT    1024
#define TL    1022                 /* T - L */
#define NROWS (256 * 1022)
#define NTILES (NROWS / 16)        /* 16352 */

#define WBUF_HALVES   65536        /* 128 KB: max phase weight footprint */
#define ACT_PER_WAVE  5120         /* bufM 16x64 + two 16x128 f16 buffers */
#define LDS_BYTES     ((WBUF_HALVES + WAVES * ACT_PER_WAVE) * 2)  /* 212992 */

// Branchless tanh: v_exp_f32 + v_rcp_f32, no EXEC-divergent paths.
__device__ __forceinline__ float fast_tanh(float x) {
  const float ax = __builtin_fabsf(x);
  const float e  = __expf(-2.0f * ax);
  const float r  = (1.0f - e) * __builtin_amdgcn_rcpf(1.0f + e);
  return __builtin_copysignf(r, x);
}

// ---- fragment loader: row-major f16 matrix -> 16x32 WMMA fragment ----
// A-matrix layout (ISA 7.12.2): lane<16 holds K=kbase+0..7 and kbase+16..23,
// lane>=16 holds K=kbase+8..15 and kbase+24..31; M (or N for B) = lane&15.
__device__ __forceinline__ v16h load_frag(const _Float16* base, int row,
                                          int rowStride, int kbase, int koff) {
  const _Float16* p = base + row * rowStride + kbase + koff;
  v8h lo = *(const v8h*)(p);
  v8h hi = *(const v8h*)(p + 16);
  v16h f;
#pragma unroll
  for (int i = 0; i < 8; ++i) { f[i] = lo[i]; f[i + 8] = hi[i]; }
  return f;
}

// One dense layer for a 16-row tile: D(16xNOUT) = A(16xK) * W^T + bias.
// src: LDS 16xK f16 row-major.  W: LDS NOUTxK f16 row-major (torch (out,in)).
// ACT: 0 = none -> outreg (C layout), 1 = tanh, 2 = relu, 3 = leaky(0.2).
template <int K, int NOUT, int ACT>
__device__ __forceinline__ void layer(const _Float16* src, const _Float16* W,
                                      const float* __restrict__ bias,
                                      _Float16* dst, float* outreg, int lane) {
  const int col   = lane & 15;
  const int mbase = (lane < 16) ? 0 : 8;
  const int koff  = (lane < 16) ? 0 : 8;
  v16h afrag[K / 32];
#pragma unroll
  for (int kc = 0; kc < K / 32; ++kc)
    afrag[kc] = load_frag(src, col, K, kc * 32, koff);
#pragma unroll
  for (int nt = 0; nt < NOUT / 16; ++nt) {
    const int n = nt * 16 + col;
    const float bv = bias[n];
    v8f acc = {bv, bv, bv, bv, bv, bv, bv, bv};
#pragma unroll
    for (int kc = 0; kc < K / 32; ++kc) {
      v16h bfrag = load_frag(W, n, K, kc * 32, koff);  // B[k][n] = W[n][k]
      acc = __builtin_amdgcn_wmma_f32_16x16x32_f16(false, afrag[kc], false,
                                                   bfrag, (short)0, acc,
                                                   false, false);
    }
#pragma unroll
    for (int r = 0; r < 8; ++r) {
      float v = acc[r];
      if (ACT == 1) v = fast_tanh(v);
      if (ACT == 2) v = fmaxf(v, 0.0f);
      if (ACT == 3) v = (v >= 0.0f) ? v : 0.2f * v;
      if (ACT == 0) outreg[nt * 8 + r] = v;
      else          dst[(mbase + r) * NOUT + n] = (_Float16)v;
    }
  }
}

// cooperative f32 -> f16 weight staging into LDS (float4 -> ds_store_b64)
__device__ __forceinline__ void stage(const float* __restrict__ src,
                                      _Float16* dst, int count) {
  for (int i = threadIdx.x * 4; i < count; i += TPB * 4) {
    const float4 f = *(const float4*)(src + i);
    v4h h;
    h[0] = (_Float16)f.x; h[1] = (_Float16)f.y;
    h[2] = (_Float16)f.z; h[3] = (_Float16)f.w;
    *(v4h*)(dst + i) = h;
  }
}

__global__ void __launch_bounds__(TPB, 1)
pnl_fused_kernel(const float* __restrict__ x,
                 const float* __restrict__ f1_hW, const float* __restrict__ f1_hb,
                 const float* __restrict__ f1_oW, const float* __restrict__ f1_ob,
                 const float* __restrict__ sW0, const float* __restrict__ sb0,
                 const float* __restrict__ sW1, const float* __restrict__ sb1,
                 const float* __restrict__ sW2, const float* __restrict__ sb2,
                 const float* __restrict__ tW0, const float* __restrict__ tb0,
                 const float* __restrict__ tW1, const float* __restrict__ tb1,
                 const float* __restrict__ tW2, const float* __restrict__ tb2,
                 float* __restrict__ resid, float* __restrict__ ldrow) {
  extern __shared__ _Float16 smem[];
  _Float16* wbuf = smem;                       // phase weights (<=65536 halves)
  _Float16* act  = smem + WBUF_HALVES;
  const int lane  = threadIdx.x & 31;
  const int wave  = threadIdx.x >> 5;
  _Float16* bufM = act + wave * ACT_PER_WAVE;  // 16x64  f16 (masked input mx)
  _Float16* bufA = bufM + 1024;                // 16x128 f16
  _Float16* bufB = bufA + 2048;                // 16x128 f16

  const int tile  = blockIdx.x * WAVES + wave;
  const int r0    = tile * 16;
  const int col   = lane & 15;
  const int mbase = (lane < 16) ? 0 : 8;

  // Load u = x[b, t+2, :] for this wave's 16 rows, in C-fragment layout:
  // lane holds column d = nt*16+col for rows m = r + mbase.
  float u_[4][8];
  const float* xbase[8];
#pragma unroll
  for (int r = 0; r < 8; ++r) {
    const int row = r0 + mbase + r;
    const int bb  = row / TL;
    const int tt  = row - bb * TL;
    xbase[r] = x + ((size_t)bb * TT + tt) * DD;  // -> x[b, t, 0]
#pragma unroll
    for (int nt = 0; nt < 4; ++nt)
      u_[nt][r] = xbase[r][2 * DD + nt * 16 + col];
  }

  float ldacc[8];
#pragma unroll
  for (int r = 0; r < 8; ++r) ldacc[r] = 0.0f;

  float s_[32], t_[32];

  // ---------------- RealNVP flow: 3 coupling blocks ----------------
  for (int blk = 0; blk < 3; ++blk) {
    __syncthreads();  // everyone done with previous wbuf contents
    stage(sW0 + blk * HH * DD, wbuf + 0,     HH * DD);
    stage(sW1 + blk * HH * HH, wbuf + 8192,  HH * HH);
    stage(sW2 + blk * DD * HH, wbuf + 24576, DD * HH);
    stage(tW0 + blk * HH * DD, wbuf + 32768, HH * DD);
    stage(tW1 + blk * HH * HH, wbuf + 40960, HH * HH);
    stage(tW2 + blk * DD * HH, wbuf + 57344, DD * HH);
    __syncthreads();

    // mx = u * m  (mask m_blk(d) = (d+blk)&1), written once into bufM
#pragma unroll
    for (int nt = 0; nt < 4; ++nt)
#pragma unroll
      for (int r = 0; r < 8; ++r) {
        const int d = nt * 16 + col;
        const float v = ((d + blk) & 1) ? u_[nt][r] : 0.0f;
        bufM[(mbase + r) * DD + d] = (_Float16)v;
      }
    // s-net: tanh, tanh, linear
    layer<DD, HH, 1>(bufM, wbuf + 0,     sb0 + blk * HH, bufA, nullptr, lane);
    layer<HH, HH, 1>(bufA, wbuf + 8192,  sb1 + blk * HH, bufB, nullptr, lane);
    layer<HH, DD, 0>(bufB, wbuf + 24576, sb2 + blk * DD, nullptr, s_, lane);
    // t-net: relu, relu, linear
    layer<DD, HH, 2>(bufM, wbuf + 32768, tb0 + blk * HH, bufA, nullptr, lane);
    layer<HH, HH, 2>(bufA, wbuf + 40960, tb1 + blk * HH, bufB, nullptr, lane);
    layer<HH, DD, 0>(bufB, wbuf + 57344, tb2 + blk * DD, nullptr, t_, lane);

    // coupling update (branchless): masked lanes keep u; others
    // u = (u - t) * exp(-s), logdet -= s
#pragma unroll
    for (int nt = 0; nt < 4; ++nt)
#pragma unroll
      for (int r = 0; r < 8; ++r) {
        const int d = nt * 16 + col;
        const bool keep = ((d + blk) & 1) != 0;
        const float sv = s_[nt * 8 + r];
        const float un = (u_[nt][r] - t_[nt * 8 + r]) * __expf(-sv);
        u_[nt][r] = keep ? u_[nt][r] : un;
        ldacc[r]  = keep ? ldacc[r]  : (ldacc[r] - sv);
      }
  }
  // u_ now holds x_inv (C layout)

  // ---------------- f1 sliding-window MLP ----------------
  __syncthreads();
  stage(f1_hW, wbuf + 0,     3 * HH * HH);
  stage(f1_oW, wbuf + 49152, DD * HH);
  __syncthreads();

  {
    // win[m] = [ x[b,t,:] | x[b,t+1,:] ]: lanes 0-15 write cols 0..63 (x[t]),
    // lanes 16-31 write cols 64..127 (x[t+1]); row m = lane&15.
    const int m   = lane & 15;
    const int row = r0 + m;
    const int bb  = row / TL;
    const int tt  = row - bb * TL;
    const float* src = x + ((size_t)bb * TT + tt + ((lane < 16) ? 0 : 1)) * DD;
    _Float16* dst = bufA + m * HH + ((lane < 16) ? 0 : DD);
#pragma unroll
    for (int i = 0; i < 16; ++i) {
      const float4 f = ((const float4*)src)[i];
      v4h h;
      h[0] = (_Float16)f.x; h[1] = (_Float16)f.y;
      h[2] = (_Float16)f.z; h[3] = (_Float16)f.w;
      *(v4h*)(dst + i * 4) = h;
    }
  }

  float h_[32];
  layer<HH, HH, 3>(bufA, wbuf + 0,     f1_hb + 0,      bufB, nullptr, lane);
  layer<HH, HH, 3>(bufB, wbuf + 16384, f1_hb + HH,     bufA, nullptr, lane);
  layer<HH, HH, 3>(bufA, wbuf + 32768, f1_hb + 2 * HH, bufB, nullptr, lane);
  layer<HH, DD, 0>(bufB, wbuf + 49152, f1_ob,          nullptr, h_, lane);

  // residuals = MLP(win) - x_inv  (both already in C layout)
#pragma unroll
  for (int nt = 0; nt < 4; ++nt)
#pragma unroll
    for (int r = 0; r < 8; ++r) {
      const size_t row = (size_t)(r0 + mbase + r);
      resid[row * DD + nt * 16 + col] = h_[nt * 8 + r] - u_[nt][r];
    }

  // per-row logdet: deterministic xor-reduce across the 16 lanes of each half
#pragma unroll
  for (int r = 0; r < 8; ++r) {
    float v = ldacc[r];
    v += __shfl_xor(v, 1, 16);
    v += __shfl_xor(v, 2, 16);
    v += __shfl_xor(v, 4, 16);
    v += __shfl_xor(v, 8, 16);
    ldacc[r] = v;
  }
  if (col == 0) {
#pragma unroll
    for (int r = 0; r < 8; ++r)
      ldrow[r0 + mbase + r] = ldacc[r];
  }
}

// Deterministic fixed-order reduction of per-row logdet over time per batch b.
__global__ void logdet_reduce_kernel(const float* __restrict__ ldrow,
                                     float* __restrict__ out) {
  __shared__ float sm[TPB];
  const int b = blockIdx.x;
  const float* p = ldrow + (size_t)b * TL;
  float acc = 0.0f;
  for (int t = threadIdx.x; t < TL; t += TPB) acc += p[t];
  sm[threadIdx.x] = acc;
  __syncthreads();
  for (int s = TPB / 2; s > 0; s >>= 1) {
    if (threadIdx.x < s) sm[threadIdx.x] += sm[threadIdx.x + s];
    __syncthreads();
  }
  if (threadIdx.x == 0) out[b] = sm[0];
}

extern "C" void kernel_launch(void* const* d_in, const int* in_sizes, int n_in,
                              void* d_out, int out_size, void* d_ws, size_t ws_size,
                              hipStream_t stream) {
  (void)in_sizes; (void)n_in; (void)out_size; (void)ws_size;
  const float* x     = (const float*)d_in[0];
  const float* f1_hW = (const float*)d_in[1];
  const float* f1_hb = (const float*)d_in[2];
  const float* f1_oW = (const float*)d_in[3];
  const float* f1_ob = (const float*)d_in[4];
  const float* sW0   = (const float*)d_in[5];
  const float* sb0   = (const float*)d_in[6];
  const float* sW1   = (const float*)d_in[7];
  const float* sb1   = (const float*)d_in[8];
  const float* sW2   = (const float*)d_in[9];
  const float* sb2   = (const float*)d_in[10];
  const float* tW0   = (const float*)d_in[11];
  const float* tb0   = (const float*)d_in[12];
  const float* tW1   = (const float*)d_in[13];
  const float* tb1   = (const float*)d_in[14];
  const float* tW2   = (const float*)d_in[15];
  const float* tb2   = (const float*)d_in[16];

  float* resid = (float*)d_out;
  float* ldout = resid + (size_t)NROWS * DD;  // output tuple tail: (B,) logdet
  float* ldrow = (float*)d_ws;                // N per-row logdet scratch

  pnl_fused_kernel<<<dim3(NTILES / WAVES), dim3(TPB), LDS_BYTES, stream>>>(
      x, f1_hW, f1_hb, f1_oW, f1_ob, sW0, sb0, sW1, sb1, sW2, sb2,
      tW0, tb0, tW1, tb1, tW2, tb2, resid, ldrow);

  logdet_reduce_kernel<<<dim3(256), dim3(TPB), 0, stream>>>(ldrow, ldout);
}
